// NegSamplerMiniBatch_48576080117775
// MI455X (gfx1250) — matrix operand order repaired and checked by reading, not compiled, not run on hardware
//
#include <hip/hip_runtime.h>
#include <cstdint>
#include <cstddef>

typedef __bf16          v16bf  __attribute__((ext_vector_type(16)));
typedef float           v8f    __attribute__((ext_vector_type(8)));
typedef unsigned short  u16x16 __attribute__((ext_vector_type(16)));

#define NROWS 65536
#define DIM   512
#define KC    256

static __device__ __forceinline__ unsigned short f32_to_bf16(float x) {
    unsigned int u = __float_as_uint(x);
    u += 0x7FFFu + ((u >> 16) & 1u);          // round-to-nearest-even
    return (unsigned short)(u >> 16);
}
static __device__ __forceinline__ float bf16_bits_to_f32(unsigned short h) {
    return __uint_as_float(((unsigned int)h) << 16);
}

// ---------------------------------------------------------------------------
// Prep: split centroids into bf16 hi/lo planes, compute ||c||^2.
// ---------------------------------------------------------------------------
__global__ void prep_kernel(const float* __restrict__ cent,
                            unsigned short* __restrict__ c_hi,
                            unsigned short* __restrict__ c_lo,
                            float* __restrict__ c_norm) {
    const int k = blockIdx.x;
    const int t = threadIdx.x;            // 0..63
    const float* row = cent + (size_t)k * DIM;
    float ss = 0.f;
#pragma unroll
    for (int j = 0; j < 8; ++j) {
        int d = t * 8 + j;
        float x = row[d];
        unsigned short hb = f32_to_bf16(x);
        float r = x - bf16_bits_to_f32(hb);
        c_hi[(size_t)k * DIM + d] = hb;
        c_lo[(size_t)k * DIM + d] = f32_to_bf16(r);
        ss += x * x;
    }
#pragma unroll
    for (int m = 16; m >= 1; m >>= 1) ss += __shfl_xor(ss, m, 32);
    __shared__ float s2[2];
    if ((t & 31) == 0) s2[t >> 5] = ss;
    __syncthreads();
    if (t == 0) c_norm[k] = s2[0] + s2[1];
}

// ---------------------------------------------------------------------------
// Main kernel: bf16x2-split WMMA distance ranking + top-2 + gather.
// Block = 256 threads (8 waves) handles 128 rows.
// ---------------------------------------------------------------------------
__global__ __launch_bounds__(256) void gemm_top2_kernel(
    const float* __restrict__ emb,
    const float* __restrict__ cent,
    const unsigned short* __restrict__ c_hi,
    const unsigned short* __restrict__ c_lo,
    const float* __restrict__ c_norm,
    float* __restrict__ out) {

    // double-buffered B slab: 2 bufs * (hi+lo) * 16 tiles * 32 lanes * 16 bf16
    __shared__ __attribute__((aligned(32))) unsigned short ldsb[2 * 2 * 16 * 32 * 16];

    const int tid   = threadIdx.x;
    const int lane  = tid & 31;
    const int wave  = tid >> 5;
    const int hl    = lane >> 4;           // lane half
    const int ln16  = lane & 15;
    const int rowBase = blockIdx.x * 128 + wave * 16;

    // Async cooperative fill of one 32-dim slab of all 256 centroids into LDS,
    // pre-swizzled into the per-lane WMMA B-fragment layout:
    //   frag(ct): lane L holds column n = 16*ct + L%16, K = 16*(L/16)+0..15.
    // Uses GLOBAL_LOAD_ASYNC_TO_LDS_B128 (ASYNCcnt) — per-lane LDS dest addr.
    auto fill_async = [&](int kb, int buf) {
        const int base = buf * 16384;      // ushort units
#pragma unroll
        for (int part = 0; part < 2; ++part) {
            const unsigned short* __restrict__ src = part ? c_lo : c_hi;
#pragma unroll
            for (int r = 0; r < 2; ++r) {
                int unit = tid * 2 + r;            // 0..511 (32B units)
                int ct   = unit >> 5;
                int ul   = unit & 31;
                int n    = ct * 16 + (ul & 15);
                int d    = kb * 32 + 16 * (ul >> 4);
                unsigned long long gaddr =
                    (unsigned long long)(uintptr_t)(src + (size_t)n * DIM + d);
                unsigned lds_a =
                    (unsigned)(uintptr_t)(&ldsb[base + part * 8192 + ct * 512 + ul * 16]);
                asm volatile("global_load_async_to_lds_b128 %0, %1, off"
                             :: "v"(lds_a), "v"(gaddr) : "memory");
                asm volatile("global_load_async_to_lds_b128 %0, %1, off offset:16"
                             :: "v"(lds_a), "v"(gaddr) : "memory");
            }
        }
    };

    v8f acc[16] = {};

    const float* __restrict__ erow = emb + (size_t)(rowBase + ln16) * DIM;

    fill_async(0, 0);

    for (int kb = 0; kb < 16; ++kb) {
        asm volatile("s_wait_asynccnt 0x0" ::: "memory");  // my fills landed
        __syncthreads();                                   // everyone's landed
        if (kb + 1 < 16) fill_async(kb + 1, (kb + 1) & 1);

        // A fragment: lane L = row L%16; K = 16*hl + 0..15
        const int d0 = kb * 32 + 16 * hl;
        float4 f0 = *reinterpret_cast<const float4*>(erow + d0);
        float4 f1 = *reinterpret_cast<const float4*>(erow + d0 + 4);
        float4 f2 = *reinterpret_cast<const float4*>(erow + d0 + 8);
        float4 f3 = *reinterpret_cast<const float4*>(erow + d0 + 12);
        float av[16] = {f0.x, f0.y, f0.z, f0.w, f1.x, f1.y, f1.z, f1.w,
                        f2.x, f2.y, f2.z, f2.w, f3.x, f3.y, f3.z, f3.w};
        u16x16 ahi_u, alo_u;
#pragma unroll
        for (int e = 0; e < 16; ++e) {
            unsigned short hb = f32_to_bf16(av[e]);
            ahi_u[e] = hb;
            alo_u[e] = f32_to_bf16(av[e] - bf16_bits_to_f32(hb));
        }
        v16bf ahi = __builtin_bit_cast(v16bf, ahi_u);
        v16bf alo = __builtin_bit_cast(v16bf, alo_u);

        const int bb = (kb & 1) * 16384;
        // Interleave tile pairs so consecutive WMMAs hit independent
        // accumulators (breaks the D->C RAW stall chain).
#pragma unroll
        for (int ct = 0; ct < 16; ct += 2) {
            u16x16 bh0_u = *reinterpret_cast<const u16x16*>(&ldsb[bb + ct * 512 + lane * 16]);
            u16x16 bl0_u = *reinterpret_cast<const u16x16*>(&ldsb[bb + 8192 + ct * 512 + lane * 16]);
            u16x16 bh1_u = *reinterpret_cast<const u16x16*>(&ldsb[bb + (ct + 1) * 512 + lane * 16]);
            u16x16 bl1_u = *reinterpret_cast<const u16x16*>(&ldsb[bb + 8192 + (ct + 1) * 512 + lane * 16]);
            v16bf bh0 = __builtin_bit_cast(v16bf, bh0_u);
            v16bf bl0 = __builtin_bit_cast(v16bf, bl0_u);
            v16bf bh1 = __builtin_bit_cast(v16bf, bh1_u);
            v16bf bl1 = __builtin_bit_cast(v16bf, bl1_u);
            acc[ct]     = __builtin_amdgcn_wmma_f32_16x16x32_bf16(false, ahi, false, bh0, (short)0, acc[ct],     false, false);
            acc[ct + 1] = __builtin_amdgcn_wmma_f32_16x16x32_bf16(false, ahi, false, bh1, (short)0, acc[ct + 1], false, false);
            acc[ct]     = __builtin_amdgcn_wmma_f32_16x16x32_bf16(false, ahi, false, bl0, (short)0, acc[ct],     false, false);
            acc[ct + 1] = __builtin_amdgcn_wmma_f32_16x16x32_bf16(false, ahi, false, bl1, (short)0, acc[ct + 1], false, false);
            acc[ct]     = __builtin_amdgcn_wmma_f32_16x16x32_bf16(false, alo, false, bh0, (short)0, acc[ct],     false, false);
            acc[ct + 1] = __builtin_amdgcn_wmma_f32_16x16x32_bf16(false, alo, false, bh1, (short)0, acc[ct + 1], false, false);
        }
    }

    // ---- rank by ||c||^2 - 2*dot (||e||^2 is row-constant, drops out) ----
    float cn[16];
#pragma unroll
    for (int ct = 0; ct < 16; ++ct) cn[ct] = c_norm[ct * 16 + ln16];

    int sec[8];
#pragma unroll
    for (int v = 0; v < 8; ++v) {          // row m = v + 8*hl
        float b1 = 3.4e38f, b2 = 3.4e38f;
        int   i1 = 0x7fffffff, i2 = 0x7fffffff;
#pragma unroll
        for (int ct = 0; ct < 16; ++ct) {
            float s = cn[ct] - 2.0f * acc[ct][v];
            int   n = ct * 16 + ln16;
            if ((s < b1) || (s == b1 && n < i1)) { b2 = b1; i2 = i1; b1 = s; i1 = n; }
            else if ((s < b2) || (s == b2 && n < i2)) { b2 = s; i2 = n; }
        }
#pragma unroll
        for (int mask = 1; mask < 16; mask <<= 1) {
            float ob1 = __shfl_xor(b1, mask, 32);
            int   oi1 = __shfl_xor(i1, mask, 32);
            float ob2 = __shfl_xor(b2, mask, 32);
            int   oi2 = __shfl_xor(i2, mask, 32);
            if ((ob1 < b1) || (ob1 == b1 && oi1 < i1)) { b2 = b1; i2 = i1; b1 = ob1; i1 = oi1; }
            else if ((ob1 < b2) || (ob1 == b2 && oi1 < i2)) { b2 = ob1; i2 = oi1; }
            if ((ob2 < b1) || (ob2 == b1 && oi2 < i1)) { b2 = b1; i2 = i1; b1 = ob2; i1 = oi2; }
            else if ((ob2 < b2) || (ob2 == b2 && oi2 < i2)) { b2 = ob2; i2 = oi2; }
        }
        sec[v] = i2;                       // 2nd-nearest centroid index
    }

    // ---- gather centroids[i2] -> out, coalesced float4 copies ----
#pragma unroll
    for (int m = 0; m < 16; ++m) {
        int idx = __shfl(sec[m & 7], (m >> 3) * 16, 32);
        const float4* __restrict__ src = reinterpret_cast<const float4*>(cent + (size_t)idx * DIM);
        float4* __restrict__ dst = reinterpret_cast<float4*>(out + (size_t)(rowBase + m) * DIM);
#pragma unroll
        for (int j = 0; j < 4; ++j)
            dst[lane + 32 * j] = src[lane + 32 * j];
    }
}

// ---------------------------------------------------------------------------
extern "C" void kernel_launch(void* const* d_in, const int* in_sizes, int n_in,
                              void* d_out, int out_size, void* d_ws, size_t ws_size,
                              hipStream_t stream) {
    const float* emb  = (const float*)d_in[0];   // (65536, 512) f32
    const float* cent = (const float*)d_in[1];   // (256, 512) f32
    // d_in[2] = batch_id (unused)

    unsigned short* c_hi   = (unsigned short*)d_ws;          // 256*512 bf16
    unsigned short* c_lo   = c_hi + (size_t)KC * DIM;        // 256*512 bf16
    float*          c_norm = (float*)(c_lo + (size_t)KC * DIM); // 256 f32
    float* out = (float*)d_out;

    hipLaunchKernelGGL(prep_kernel, dim3(KC), dim3(64), 0, stream,
                       cent, c_hi, c_lo, c_norm);
    hipLaunchKernelGGL(gemm_top2_kernel, dim3(NROWS / 128), dim3(256), 0, stream,
                       emb, cent, c_hi, c_lo, c_norm, out);
}
